// MixtureOfSoftmax_24215025615519
// MI455X (gfx1250) — matrix operand order repaired and checked by reading, not compiled, not run on hardware
//
#include <hip/hip_runtime.h>
#include <hip/hip_bf16.h>
#include <math.h>

typedef __attribute__((ext_vector_type(16))) _Float16 v16h;
typedef __attribute__((ext_vector_type(8)))  _Float16 v8h;
typedef __attribute__((ext_vector_type(8)))  float    v8f;

#define DEVINL __device__ __forceinline__

#if __has_builtin(__builtin_amdgcn_sched_barrier)
#define SCHED_FENCE() __builtin_amdgcn_sched_barrier(0)
#else
#define SCHED_FENCE() asm volatile("" ::: "memory")
#endif

constexpr int kS     = 2048;          // sequence
constexpr int kD     = 1024;          // hidden dim
constexpr int kMix   = 2;             // mixtures
constexpr int kV     = 50257;         // vocab
constexpr int kVpad  = 50432;         // 197 * 256 (multiple of block-N)
constexpr int kM     = kS * kMix;     // 4096 rows of big GEMM
constexpr int kNChunk = kVpad / 256;  // 197 N-chunks

// LDS B tile: [2 buffers][256 N rows][72 halfs]. Row stride 144B = 36 banks:
// 16-lane ds_load_b128 starts hit banks 36*r % 64 = {0,4,...,60} -- an exact
// 64-bank partition, conflict-free (also for the +16B hi halves).
// 2 x 36KB = 72KB LDS per block.
constexpr int kBRow = 72;

// ---------- WMMA helpers (layouts per CDNA5 ISA 7.12.2, wave32) ----------

// A fragment 16x32 f16: lane L -> M = L%16; K = 8*(L/16) + {0..7} and +16.
DEVINL v16h load_a_frag(const _Float16* p) {
  v8h lo = *(const v8h*)(p);
  v8h hi = *(const v8h*)(p + 16);
  return __builtin_shufflevector(lo, hi, 0, 1, 2, 3, 4, 5, 6, 7,
                                 8, 9, 10, 11, 12, 13, 14, 15);
}

// B fragment slice from LDS: two 16B chunks at K offsets 0 and +16, 16B-aligned (144B rows).
DEVINL v16h load_b_lds(const _Float16* p) {
  v8h lo = *(const v8h*)(p);
  v8h hi = *(const v8h*)(p + 16);
  return __builtin_shufflevector(lo, hi, 0, 1, 2, 3, 4, 5, 6, 7,
                                 8, 9, 10, 11, 12, 13, 14, 15);
}

DEVINL v8f wmma_f16(v16h a, v16h b, v8f c) {
  // emits v_wmma_f32_16x16x32_f16
  return __builtin_amdgcn_wmma_f32_16x16x32_f16(
      /*neg_a=*/false, a, /*neg_b=*/false, b,
      /*c_mod=*/(short)0, c, /*reuse_a=*/false, /*reuse_b=*/false);
}

// Stage one 256(N) x 64(K) f16 B tile slice into LDS, one 128B row per thread,
// via async global->LDS DMA (ASYNCcnt). IOFFSET is added to both the global
// and LDS addresses per the ISA pseudocode, so one address pair serves all 8.
DEVINL void stage_b_async(const _Float16* __restrict__ W, int ldw,
                          int n_blk, int k, _Float16 (*Bs)[kBRow]) {
  const int i = threadIdx.x;  // 0..255 -> one B row each
  const _Float16* g = W + (size_t)(n_blk + i) * ldw + k;
  unsigned int l = (unsigned int)(uintptr_t)(&Bs[i][0]);
  asm volatile(
      "global_load_async_to_lds_b128 %0, %1, off\n\t"
      "global_load_async_to_lds_b128 %0, %1, off offset:16\n\t"
      "global_load_async_to_lds_b128 %0, %1, off offset:32\n\t"
      "global_load_async_to_lds_b128 %0, %1, off offset:48\n\t"
      "global_load_async_to_lds_b128 %0, %1, off offset:64\n\t"
      "global_load_async_to_lds_b128 %0, %1, off offset:80\n\t"
      "global_load_async_to_lds_b128 %0, %1, off offset:96\n\t"
      "global_load_async_to_lds_b128 %0, %1, off offset:112"
      :
      : "v"(l), "v"(g)
      : "memory");
}

// Block-cooperative GEMM: block tile 64(M) x 256(N), 8 waves as 2(Mgrp)x4(Nsub),
// each wave accumulating a 32x64 f32 tile in c[2][4].
// A: row-major [*, lda] f16 (rows = M), W: row-major [*, ldw] f16 (rows = N),
// C = A @ W^T over kD. B tile double-buffered (K=64 deep) in LDS via async DMA.
DEVINL void gemm_block(const _Float16* __restrict__ A, int lda,
                       const _Float16* __restrict__ W, int ldw,
                       int m0, int n_blk, _Float16 (*Bs)[kBRow], v8f c[2][4]) {
  const int lane = threadIdx.x & 31;
  const int half = lane >> 4;
  const int l15  = lane & 15;
  const int nsub = (threadIdx.x >> 5) & 3;
  const _Float16* a0 = A + (size_t)(m0 + l15) * lda + (half << 3);
  const _Float16* a1 = a0 + (size_t)16 * lda;

  stage_b_async(W, ldw, n_blk, 0, Bs);  // prologue: fill buffer 0
  for (int k = 0; k < kD; k += 64) {
    const int cur = (k >> 6) & 1;
    if (k + 64 < kD) {
      stage_b_async(W, ldw, n_blk, k + 64, Bs + ((cur ^ 1) << 8));
      asm volatile("s_wait_asynccnt 8" ::: "memory");  // prev stage's 8 done
    } else {
      asm volatile("s_wait_asynccnt 0" ::: "memory");
    }
    __syncthreads();  // all waves' DMAs for 'cur' are visible

    const _Float16(*Bc)[kBRow] = Bs + (cur << 8);
#pragma unroll
    for (int ks = 0; ks < 64; ks += 32) {
      __builtin_prefetch(a0 + k + ks + 256, 0, 3);  // global_prefetch_b8
      v16h af0 = load_a_frag(a0 + k + ks);
      v16h af1 = load_a_frag(a1 + k + ks);
      v16h bf[4];
#pragma unroll
      for (int t = 0; t < 4; ++t)  // one clause of 8 ds_load_b128
        bf[t] = load_b_lds(&Bc[nsub * 64 + t * 16 + l15][ks + half * 8]);
      SCHED_FENCE();  // keep all loads above, all WMMAs below (partial waits)
#pragma unroll
      for (int t = 0; t < 4; ++t) {  // 8 back-to-back WMMAs, no LDS stalls
        c[0][t] = wmma_f16(af0, bf[t], c[0][t]);
        c[1][t] = wmma_f16(af1, bf[t], c[1][t]);
      }
    }
    __syncthreads();  // reads of 'cur' done before it is restaged next iter
  }
}

// ---------- Kernel 1: fp32 -> f16 conversion with tail zero-pad ----------
__global__ void __launch_bounds__(256)
convert_kernel(const float* __restrict__ src, _Float16* __restrict__ dst,
               size_t n_total, size_t n_src) {
  size_t i = (size_t)blockIdx.x * blockDim.x + threadIdx.x;
  size_t stride = (size_t)gridDim.x * blockDim.x;
  for (; i < n_total; i += stride)
    dst[i] = (i < n_src) ? (_Float16)src[i] : (_Float16)0.0f;
}

// ---------- Kernel 2: gate logits + log(pi + 1e-10), one wave per row ----
__global__ void __launch_bounds__(256)
gate_kernel(const float* __restrict__ hidden, const float* __restrict__ w_gate,
            float* __restrict__ log_pi) {
  int s    = (int)((blockIdx.x * blockDim.x + threadIdx.x) >> 5);
  int lane = threadIdx.x & 31;
  if (s >= kS) return;
  const float* h = hidden + (size_t)s * kD;
  float g0 = 0.f, g1 = 0.f;
  for (int d = lane; d < kD; d += 32) {
    float hv = h[d];
    g0 += hv * w_gate[d];
    g1 += hv * w_gate[kD + d];
  }
#pragma unroll
  for (int d = 16; d >= 1; d >>= 1) {
    g0 += __shfl_xor(g0, d, 32);
    g1 += __shfl_xor(g1, d, 32);
  }
  if (lane == 0) {
    float mx = fmaxf(g0, g1);
    float l2 = mx + logf(expf(g0 - mx) + expf(g1 - mx));
    log_pi[s * 2 + 0] = logf(expf(g0 - l2) + 1e-10f);
    log_pi[s * 2 + 1] = logf(expf(g1 - l2) + 1e-10f);
  }
}

// ---------- Kernel 3: projected = hidden @ w_proj^T  (f16 out) ------------
__global__ void __launch_bounds__(256)
proj_gemm_kernel(const _Float16* __restrict__ A,  // hidden_h [2048 x 1024]
                 const _Float16* __restrict__ W,  // wproj_h  [2048 x 1024]
                 _Float16* __restrict__ P) {      // proj_h   [2048 x 2048]
  __shared__ __align__(16) _Float16 Bs[512][kBRow];
  const int w    = threadIdx.x >> 5;
  const int lane = threadIdx.x & 31;
  const int half = lane >> 4;
  const int l15  = lane & 15;
  const int m0    = blockIdx.x * 64 + (w >> 2) * 32;
  const int n_blk = blockIdx.y * 256;
  const int n0    = n_blk + (w & 3) * 64;
  v8f c[2][4] = {};
  gemm_block(A, kD, W, kD, m0, n_blk, Bs, c);
#pragma unroll
  for (int msub = 0; msub < 2; ++msub)
#pragma unroll
    for (int t = 0; t < 4; ++t)
#pragma unroll
      for (int r = 0; r < 8; ++r) {
        int m = m0 + msub * 16 + half * 8 + r;
        int n = n0 + t * 16 + l15;
        P[(size_t)m * (kMix * kD) + n] = (_Float16)c[msub][t][r];
      }
}

// ---------- Kernel 4: logits tile -> per-(row, chunk) (max, sumexp) -------
__global__ void __launch_bounds__(256)
lse_partial_kernel(const _Float16* __restrict__ A,  // proj_h as [4096 x 1024]
                   const _Float16* __restrict__ W,  // weight_h [Vpad x 1024]
                   float2* __restrict__ partials) { // [4096 x kNChunk]
  __shared__ __align__(16) _Float16 Bs[512][kBRow];
  __shared__ float2 stats[8][32];
  const int w    = threadIdx.x >> 5;
  const int lane = threadIdx.x & 31;
  const int half = lane >> 4;
  const int l15  = lane & 15;
  const int m0    = blockIdx.x * 64 + (w >> 2) * 32;
  const int n_blk = blockIdx.y * 256;
  const int n0    = n_blk + (w & 3) * 64;
  v8f c[2][4] = {};
  gemm_block(A, kD, W, kD, m0, n_blk, Bs, c);
#pragma unroll
  for (int msub = 0; msub < 2; ++msub) {
#pragma unroll
    for (int r = 0; r < 8; ++r) {
      float mx = -INFINITY;
#pragma unroll
      for (int t = 0; t < 4; ++t) {
        int v = n0 + t * 16 + l15;
        float x = (v < kV) ? c[msub][t][r] : -INFINITY;
        mx = fmaxf(mx, x);
      }
#pragma unroll
      for (int d = 8; d >= 1; d >>= 1) mx = fmaxf(mx, __shfl_xor(mx, d, 16));
      float sm = 0.0f;
#pragma unroll
      for (int t = 0; t < 4; ++t) {
        int v = n0 + t * 16 + l15;
        if (v < kV) sm += __expf(c[msub][t][r] - mx);
      }
#pragma unroll
      for (int d = 8; d >= 1; d >>= 1) sm += __shfl_xor(sm, d, 16);
      if (l15 == 0) stats[w][msub * 16 + half * 8 + r] = make_float2(mx, sm);
    }
  }
  __syncthreads();
  if (threadIdx.x < 64) {  // combine the 4 waves sharing each row group
    int mg  = threadIdx.x >> 5;
    int row = threadIdx.x & 31;
    float M = -INFINITY, S = 0.0f;
#pragma unroll
    for (int q = 0; q < 4; ++q) {
      float2 p = stats[mg * 4 + q][row];
      float nm = fmaxf(M, p.x);
      S = S * __expf(M - nm) + p.y * __expf(p.x - nm);
      M = nm;
    }
    int grow = blockIdx.x * 64 + mg * 32 + row;
    partials[(size_t)grow * kNChunk + blockIdx.y] = make_float2(M, S);
  }
}

// ---------- Kernel 5: reduce chunks -> lse[row], one wave per row ---------
__global__ void __launch_bounds__(256)
lse_reduce_kernel(const float2* __restrict__ partials, float* __restrict__ lse) {
  int row  = (int)((blockIdx.x * blockDim.x + threadIdx.x) >> 5);
  int lane = threadIdx.x & 31;
  if (row >= kM) return;
  float M = -INFINITY, S = 0.0f;
  for (int cidx = lane; cidx < kNChunk; cidx += 32) {
    float2 p = partials[(size_t)row * kNChunk + cidx];
    float nm = fmaxf(M, p.x);
    S = S * __expf(M - nm) + p.y * __expf(p.x - nm);
    M = nm;
  }
#pragma unroll
  for (int d = 16; d >= 1; d >>= 1) {
    float Mo = __shfl_xor(M, d, 32);
    float So = __shfl_xor(S, d, 32);
    float nm = fmaxf(M, Mo);
    S = S * __expf(M - nm) + So * __expf(Mo - nm);
    M = nm;
  }
  if (lane == 0) lse[row] = M + logf(S);
}

// ---------- Kernel 6: recompute logits, fuse log-softmax + K-mixture ------
__global__ void __launch_bounds__(256)
mix_out_kernel(const _Float16* __restrict__ A, const _Float16* __restrict__ W,
               const float* __restrict__ log_pi, const float* __restrict__ lse,
               float* __restrict__ out) {
  __shared__ __align__(16) _Float16 Bs[512][kBRow];
  const int w    = threadIdx.x >> 5;
  const int lane = threadIdx.x & 31;
  const int half = lane >> 4;
  const int l15  = lane & 15;
  const int m0    = blockIdx.x * 64 + (w >> 2) * 32;
  const int n_blk = blockIdx.y * 256;
  const int n0    = n_blk + (w & 3) * 64;
  v8f c[2][4] = {};
  gemm_block(A, kD, W, kD, m0, n_blk, Bs, c);
  // rows m and m+1 (k=0/k=1 of the same s) sit in even/odd VGPRs of one lane
#pragma unroll
  for (int msub = 0; msub < 2; ++msub) {
    const int mbase = m0 + msub * 16 + half * 8;
#pragma unroll
    for (int q = 0; q < 4; ++q) {
      const int m = mbase + 2 * q;   // even -> k=0; m+1 -> k=1
      const int s = m >> 1;
      const float lp0 = log_pi[m],     lp1 = log_pi[m + 1];
      const float z0  = lse[m],        z1  = lse[m + 1];
#pragma unroll
      for (int t = 0; t < 4; ++t) {
        int v = n0 + t * 16 + l15;
        if (v < kV) {
          float a0 = lp0 + c[msub][t][2 * q]     - z0;
          float a1 = lp1 + c[msub][t][2 * q + 1] - z1;
          float hi = fmaxf(a0, a1);
          out[(size_t)s * kV + v] = hi + log1pf(__expf(-fabsf(a0 - a1)));
        }
      }
    }
  }
}

// -------------------------------------------------------------------------
extern "C" void kernel_launch(void* const* d_in, const int* in_sizes, int n_in,
                              void* d_out, int out_size, void* d_ws, size_t ws_size,
                              hipStream_t stream) {
  const float* hidden = (const float*)d_in[0];  // [1,2048,1024]
  const float* weight = (const float*)d_in[1];  // [50257,1024]
  const float* w_proj = (const float*)d_in[2];  // [2048,1024]
  const float* w_gate = (const float*)d_in[3];  // [2,1024]
  float* out = (float*)d_out;                   // [1,2048,50257]

  char* ws = (char*)d_ws;
  size_t off = 0;
  auto take = [&](size_t bytes) -> char* {
    char* p = ws + off;
    off = (off + bytes + 255) & ~(size_t)255;
    return p;
  };
  _Float16* hidden_h = (_Float16*)take((size_t)kS * kD * 2);            //   4 MB
  _Float16* wproj_h  = (_Float16*)take((size_t)kMix * kD * kD * 2);     //   4 MB
  _Float16* weight_h = (_Float16*)take((size_t)kVpad * kD * 2);         // 103 MB
  _Float16* proj_h   = (_Float16*)take((size_t)kS * kMix * kD * 2);     //   8 MB
  float*    log_pi   = (float*)take((size_t)kM * sizeof(float));
  float*    lse      = (float*)take((size_t)kM * sizeof(float));
  float2*   partials = (float2*)take((size_t)kM * kNChunk * sizeof(float2)); // 6.5 MB

  convert_kernel<<<2048, 256, 0, stream>>>(hidden, hidden_h,
                                           (size_t)kS * kD, (size_t)kS * kD);
  convert_kernel<<<2048, 256, 0, stream>>>(w_proj, wproj_h,
                                           (size_t)kMix * kD * kD,
                                           (size_t)kMix * kD * kD);
  convert_kernel<<<4096, 256, 0, stream>>>(weight, weight_h,
                                           (size_t)kVpad * kD, (size_t)kV * kD);

  gate_kernel<<<(kS * 32 + 255) / 256, 256, 0, stream>>>(hidden, w_gate, log_pi);

  proj_gemm_kernel<<<dim3(kS / 64, (kMix * kD) / 256), 256, 0, stream>>>(
      hidden_h, wproj_h, proj_h);

  lse_partial_kernel<<<dim3(kM / 64, kNChunk), 256, 0, stream>>>(
      proj_h, weight_h, partials);

  lse_reduce_kernel<<<(kM * 32 + 255) / 256, 256, 0, stream>>>(partials, lse);

  mix_out_kernel<<<dim3(kM / 64, kNChunk), 256, 0, stream>>>(
      proj_h, weight_h, log_pi, lse, out);
}
